// BiLSTMCRF_20426864459919
// MI455X (gfx1250) — compile-verified, matrix operand order checked by reading, not run on hardware
//
#include <hip/hip_runtime.h>
#include <hip/hip_bf16.h>
#include <math.h>

// ---------------------------------------------------------------------------
// BiLSTM-CRF forward loss for MI455X (gfx1250), wave32 + WMMA bf16.
// ---------------------------------------------------------------------------

typedef __bf16 bf16_t;
typedef bf16_t v16bf __attribute__((ext_vector_type(16)));
typedef float  v8f   __attribute__((ext_vector_type(8)));

#define WMMA_BF16(a, b, c) \
    __builtin_amdgcn_wmma_f32_16x16x32_bf16(false, (a), false, (b), (short)0, (c), false, false)

constexpr int Bb  = 64;    // batch
constexpr int Ss  = 512;   // sequence
constexpr int Ee  = 256;   // embedding dim
constexpr int Hh  = 256;   // hidden per direction
constexpr int HID = 512;   // bidirectional hidden
constexpr int Kt  = 10;    // num tags

__device__ inline float sigf(float x) { return 1.0f / (1.0f + __expf(-x)); }
// tanh via exp with saturation-safe form: 1 - 2/(e^{2x}+1)
__device__ inline float tanh_fast(float x) {
    float e = __expf(2.0f * x);
    return 1.0f - 2.0f / (e + 1.0f);
}

// ---------------------------------------------------------------------------
// Weight prep: Wcat[j, 0:in) = w_ih[j,:], Wcat[j, in:in+256) = w_hh[j,:] (bf16)
// bsum[j] = b_ih[j] + b_hh[j]
// ---------------------------------------------------------------------------
__global__ void build_wcat_kernel(const float* __restrict__ w_ih,
                                  const float* __restrict__ w_hh,
                                  const float* __restrict__ b_ih,
                                  const float* __restrict__ b_hh,
                                  bf16_t* __restrict__ wcat,
                                  float* __restrict__ bsum,
                                  int in_dim) {
    int idx = blockIdx.x * blockDim.x + threadIdx.x;
    int kc = in_dim + Hh;
    int total = 4 * Hh * kc;
    if (idx < total) {
        int j = idx / kc, c = idx % kc;
        float v = (c < in_dim) ? w_ih[(size_t)j * in_dim + c]
                               : w_hh[(size_t)j * Hh + (c - in_dim)];
        wcat[idx] = (bf16_t)v;
    }
    if (idx < 4 * Hh) bsum[idx] = b_ih[idx] + b_hh[idx];
}

__global__ void build_linw_kernel(const float* __restrict__ w,
                                  bf16_t* __restrict__ lw) {
    int idx = blockIdx.x * blockDim.x + threadIdx.x;  // 16 * 512
    if (idx >= 16 * HID) return;
    int j = idx / HID, k = idx % HID;
    lw[idx] = (j < Kt) ? (bf16_t)w[(size_t)j * HID + k] : (bf16_t)0.0f;
}

// ---------------------------------------------------------------------------
// Embedding gather -> bf16 activations [B*S, E]
// ---------------------------------------------------------------------------
__global__ void embed_kernel(const int* __restrict__ tok,
                             const float* __restrict__ emb,
                             bf16_t* __restrict__ x) {
    int idx = blockIdx.x * blockDim.x + threadIdx.x;
    if (idx >= Bb * Ss * Ee) return;
    int r = idx / Ee, e = idx % Ee;
    x[idx] = (bf16_t)emb[(size_t)tok[r] * Ee + e];
}

// ---------------------------------------------------------------------------
// Persistent per-direction LSTM layer. grid=2 (dir), block=1024 (32 waves).
// Per step: z[64,1024] = [x_t | h] @ Wcat^T via WMMA; c in regs, h in LDS.
// Wave w owns batch tile m0=(w%4)*16 and unit tiles u=(w/4)+{0,8} (16 wide),
// computing all 4 gates locally so the LSTM cell update needs no exchange.
// ---------------------------------------------------------------------------
template <int IN>
__global__ __launch_bounds__(1024) void lstm_layer_kernel(
    const bf16_t* __restrict__ x_in,      // [B*S, IN]
    const bf16_t* __restrict__ wcat_all,  // [2][1024, IN+256]
    const float* __restrict__ bsum_all,   // [2][1024]
    bf16_t* __restrict__ x_out) {         // [B*S, 512]
    constexpr int KC = IN + Hh;
    constexpr int KX = IN / 32;
    constexpr int KH = Hh / 32;
    const int dir = blockIdx.x;
    const bf16_t* wcat = wcat_all + (size_t)dir * (4 * Hh) * KC;
    const float* bsum = bsum_all + dir * (4 * Hh);

    __shared__ bf16_t h_lds[Bb][Hh];  // 32 KB

    const int tid = threadIdx.x;
    const int lane = tid & 31;
    const int wv = tid >> 5;
    const int m0 = (wv & 3) * 16;
    const int ub = wv >> 2;
    const int lrow = lane & 15;
    const int khalf = (lane >> 4) * 16;

    for (int i = tid; i < Bb * Hh; i += 1024) ((bf16_t*)h_lds)[i] = (bf16_t)0.0f;

    float cc[2][8];
    float bias[2][4];
#pragma unroll
    for (int j = 0; j < 2; ++j) {
#pragma unroll
        for (int v = 0; v < 8; ++v) cc[j][v] = 0.0f;
#pragma unroll
        for (int g = 0; g < 4; ++g)
            bias[j][g] = bsum[g * Hh + (ub + j * 8) * 16 + lrow];
    }
    __syncthreads();

#pragma unroll 1
    for (int step = 0; step < Ss; ++step) {
        const int t = dir ? (Ss - 1 - step) : step;
        const int tn = dir ? (t > 0 ? t - 1 : 0) : (t < Ss - 1 ? t + 1 : t);
        // Prefetch next timestep's activation row (global_prefetch_b8).
        __builtin_prefetch(x_in + ((size_t)(m0 + lrow) * Ss + tn) * IN + khalf, 0, 0);

        float hv[2][8];
#pragma unroll
        for (int j = 0; j < 2; ++j) {
            const int u0 = (ub + j * 8) * 16;
            const v8f vz = {0.f, 0.f, 0.f, 0.f, 0.f, 0.f, 0.f, 0.f};
            v8f acc[4];
#pragma unroll
            for (int g = 0; g < 4; ++g) acc[g] = vz;

            // x_t contribution (K = IN), A from global (L2-resident bf16)
            const bf16_t* arow = x_in + ((size_t)(m0 + lrow) * Ss + t) * IN;
#pragma unroll
            for (int kk = 0; kk < KX; ++kk) {
                v16bf a = *(const v16bf*)(arow + kk * 32 + khalf);
#pragma unroll
                for (int g = 0; g < 4; ++g) {
                    const bf16_t* bp =
                        wcat + (size_t)(g * Hh + u0 + lrow) * KC + kk * 32 + khalf;
                    v16bf bb = *(const v16bf*)bp;
                    acc[g] = WMMA_BF16(a, bb, acc[g]);
                }
            }
            // h_{t-1} contribution (K = 256), A from LDS
#pragma unroll
            for (int kk = 0; kk < KH; ++kk) {
                v16bf a = *(const v16bf*)(&h_lds[m0 + lrow][kk * 32 + khalf]);
#pragma unroll
                for (int g = 0; g < 4; ++g) {
                    const bf16_t* bp =
                        wcat + (size_t)(g * Hh + u0 + lrow) * KC + IN + kk * 32 + khalf;
                    v16bf bb = *(const v16bf*)bp;
                    acc[g] = WMMA_BF16(a, bb, acc[g]);
                }
            }
            // LSTM cell update (gate order i,f,g,o), c held in registers
#pragma unroll
            for (int v = 0; v < 8; ++v) {
                float iv = sigf(acc[0][v] + bias[j][0]);
                float fv = sigf(acc[1][v] + bias[j][1]);
                float gv = tanh_fast(acc[2][v] + bias[j][2]);
                float ov = sigf(acc[3][v] + bias[j][3]);
                float cn = fv * cc[j][v] + iv * gv;
                cc[j][v] = cn;
                hv[j][v] = ov * tanh_fast(cn);
            }
        }
        __syncthreads();  // all waves done reading old h
#pragma unroll
        for (int j = 0; j < 2; ++j) {
            const int u0 = (ub + j * 8) * 16;
#pragma unroll
            for (int v = 0; v < 8; ++v) {
                int m = v + ((lane >> 4) << 3);  // D-matrix layout: M = v + 8*(lane/16)
                int b = m0 + m;
                int u = u0 + lrow;
                bf16_t hb = (bf16_t)hv[j][v];
                h_lds[b][u] = hb;
                x_out[((size_t)b * Ss + t) * HID + dir * Hh + u] = hb;
            }
        }
        __syncthreads();  // new h visible for next step
    }
}

// ---------------------------------------------------------------------------
// Emissions: em[B*S, 10] = x @ linear_w^T + b  (WMMA, N padded to 16)
// ---------------------------------------------------------------------------
__global__ void linear_em_kernel(const bf16_t* __restrict__ x,   // [B*S, 512]
                                 const bf16_t* __restrict__ lw,  // [16, 512]
                                 const float* __restrict__ lb,   // [10]
                                 float* __restrict__ em) {       // [B*S, 10]
    const int lane = threadIdx.x & 31;
    const int wv = threadIdx.x >> 5;
    const int tile = blockIdx.x * 8 + wv;
    const int m0 = tile * 16;
    const int lrow = lane & 15;
    const int khalf = (lane >> 4) * 16;
    v8f acc = {0.f, 0.f, 0.f, 0.f, 0.f, 0.f, 0.f, 0.f};
    const bf16_t* arow = x + (size_t)(m0 + lrow) * HID;
#pragma unroll
    for (int kk = 0; kk < HID / 32; ++kk) {
        v16bf a = *(const v16bf*)(arow + kk * 32 + khalf);
        v16bf b = *(const v16bf*)(lw + (size_t)lrow * HID + kk * 32 + khalf);
        acc = WMMA_BF16(a, b, acc);
    }
    if (lrow < Kt) {
        float bv = lb[lrow];
#pragma unroll
        for (int v = 0; v < 8; ++v) {
            int m = v + ((lane >> 4) << 3);
            em[(size_t)(m0 + m) * Kt + lrow] = acc[v] + bv;
        }
    }
}

// ---------------------------------------------------------------------------
// CRF: per-batch forward log-partition + gold score; K=10 scalar per lane.
// ---------------------------------------------------------------------------
__global__ void crf_kernel(const float* __restrict__ em,
                           const int* __restrict__ tokens,
                           const int* __restrict__ tags,
                           const float* __restrict__ T,
                           float* __restrict__ perb) {
    __shared__ float Tsh[Kt * Kt];
    for (int i = threadIdx.x; i < Kt * Kt; i += blockDim.x) Tsh[i] = T[i];
    __syncthreads();
    int b = threadIdx.x;
    if (b >= Bb) return;
    const float* eb = em + (size_t)b * Ss * Kt;
    const int* tokb = tokens + b * Ss;
    const int* tagb = tags + b * Ss;

    float sc[Kt];
#pragma unroll
    for (int k = 0; k < Kt; ++k) sc[k] = eb[k];
    int tp = tagb[0];
    if (tp == 9) tp = 0;
    float gold = eb[tp];

    for (int t = 1; t < Ss; ++t) {
        const float* e = eb + t * Kt;
        bool m = (tokb[t] != 0);
        float ns[Kt];
#pragma unroll
        for (int jt = 0; jt < Kt; ++jt) {
            float mx = -1e30f;
#pragma unroll
            for (int i = 0; i < Kt; ++i) mx = fmaxf(mx, sc[i] + Tsh[i * Kt + jt]);
            float s = 0.f;
#pragma unroll
            for (int i = 0; i < Kt; ++i) s += __expf(sc[i] + Tsh[i * Kt + jt] - mx);
            ns[jt] = mx + __logf(s) + e[jt];
        }
        if (m) {
#pragma unroll
            for (int k = 0; k < Kt; ++k) sc[k] = ns[k];
        }
        int tg = tagb[t];
        if (tg == 9) tg = 0;
        if (m) gold += Tsh[tp * Kt + tg] + e[tg];
        tp = tg;
    }
    float mx = -1e30f;
#pragma unroll
    for (int k = 0; k < Kt; ++k) mx = fmaxf(mx, sc[k]);
    float s = 0.f;
#pragma unroll
    for (int k = 0; k < Kt; ++k) s += __expf(sc[k] - mx);
    perb[b] = (mx + __logf(s)) - gold;
}

__global__ void reduce_kernel(const float* __restrict__ perb,
                              float* __restrict__ out) {
    __shared__ float sh[Bb];
    sh[threadIdx.x] = perb[threadIdx.x];
    __syncthreads();
    if (threadIdx.x == 0) {
        float s = 0.f;
        for (int i = 0; i < Bb; ++i) s += sh[i];
        out[0] = s / (float)Bb;
    }
}

// ---------------------------------------------------------------------------
// Launch. Input flattening (setup_inputs dict order, params nested in order):
// 0 tokens, 1 tags, 2 lengths, 3 embedding,
// 4..7  L0 fwd (w_ih,w_hh,b_ih,b_hh), 8..11 L0 bwd,
// 12..15 L1 fwd, 16..19 L1 bwd, 20 linear_w, 21 linear_b, 22 transitions
// ---------------------------------------------------------------------------
extern "C" void kernel_launch(void* const* d_in, const int* in_sizes, int n_in,
                              void* d_out, int out_size, void* d_ws, size_t ws_size,
                              hipStream_t stream) {
    (void)in_sizes; (void)n_in; (void)out_size; (void)ws_size;
    const int* tokens = (const int*)d_in[0];
    const int* tags = (const int*)d_in[1];
    const float* embedding = (const float*)d_in[3];
    const float* W[4][4];  // [layer*2+dir][w_ih,w_hh,b_ih,b_hh]
    for (int ld = 0; ld < 4; ++ld)
        for (int q = 0; q < 4; ++q) W[ld][q] = (const float*)d_in[4 + ld * 4 + q];
    const float* lin_w = (const float*)d_in[20];
    const float* lin_b = (const float*)d_in[21];
    const float* trans = (const float*)d_in[22];

    // Workspace carve-out (aligned to 256 B).
    char* ws = (char*)d_ws;
    size_t off = 0;
    auto alloc = [&](size_t bytes) -> char* {
        char* p = ws + off;
        off = (off + bytes + 255) & ~(size_t)255;
        return p;
    };
    bf16_t* XA = (bf16_t*)alloc((size_t)Bb * Ss * Ee * sizeof(bf16_t));     // embeddings
    bf16_t* XB = (bf16_t*)alloc((size_t)Bb * Ss * HID * sizeof(bf16_t));    // layer0 out
    bf16_t* XC = (bf16_t*)alloc((size_t)Bb * Ss * HID * sizeof(bf16_t));    // layer1 out
    bf16_t* WC0 = (bf16_t*)alloc((size_t)2 * 1024 * (Ee + Hh) * sizeof(bf16_t));
    bf16_t* WC1 = (bf16_t*)alloc((size_t)2 * 1024 * (HID + Hh) * sizeof(bf16_t));
    float* BS0 = (float*)alloc((size_t)2 * 1024 * sizeof(float));
    float* BS1 = (float*)alloc((size_t)2 * 1024 * sizeof(float));
    bf16_t* LINW = (bf16_t*)alloc((size_t)16 * HID * sizeof(bf16_t));
    float* EM = (float*)alloc((size_t)Bb * Ss * Kt * sizeof(float));
    float* PERB = (float*)alloc((size_t)Bb * sizeof(float));

    // --- weight prep ---
    {
        int kc0 = Ee + Hh, kc1 = HID + Hh;
        int nb0 = (4 * Hh * kc0 + 255) / 256;
        int nb1 = (4 * Hh * kc1 + 255) / 256;
        build_wcat_kernel<<<nb0, 256, 0, stream>>>(W[0][0], W[0][1], W[0][2], W[0][3],
                                                   WC0, BS0, Ee);
        build_wcat_kernel<<<nb0, 256, 0, stream>>>(W[1][0], W[1][1], W[1][2], W[1][3],
                                                   WC0 + (size_t)1024 * kc0, BS0 + 1024, Ee);
        build_wcat_kernel<<<nb1, 256, 0, stream>>>(W[2][0], W[2][1], W[2][2], W[2][3],
                                                   WC1, BS1, HID);
        build_wcat_kernel<<<nb1, 256, 0, stream>>>(W[3][0], W[3][1], W[3][2], W[3][3],
                                                   WC1 + (size_t)1024 * kc1, BS1 + 1024, HID);
        build_linw_kernel<<<(16 * HID + 255) / 256, 256, 0, stream>>>(lin_w, LINW);
    }
    // --- embedding gather ---
    embed_kernel<<<(Bb * Ss * Ee + 255) / 256, 256, 0, stream>>>(tokens, embedding, XA);
    // --- recurrent layers (persistent, one WGP per direction) ---
    lstm_layer_kernel<Ee><<<2, 1024, 0, stream>>>(XA, WC0, BS0, XB);
    lstm_layer_kernel<HID><<<2, 1024, 0, stream>>>(XB, WC1, BS1, XC);
    // --- emissions + CRF ---
    linear_em_kernel<<<(Bb * Ss / 16) / 8, 256, 0, stream>>>(XC, LINW, lin_b, EM);
    crf_kernel<<<1, 64, 0, stream>>>(EM, tokens, tags, trans, PERB);
    reduce_kernel<<<1, Bb, 0, stream>>>(PERB, (float*)d_out);
}
